// PixelLevelVotingAttention_41875931136243
// MI455X (gfx1250) — matrix operand order repaired and checked by reading, not compiled, not run on hardware
//
#include <hip/hip_runtime.h>
#include <hip/hip_bf16.h>

typedef __attribute__((ext_vector_type(16))) _Float16 v16h;
typedef __attribute__((ext_vector_type(8)))  float    v8f;

#if defined(__has_builtin)
# if __has_builtin(__builtin_amdgcn_global_load_async_to_lds_b128) && \
     __has_builtin(__builtin_amdgcn_s_wait_asynccnt)
#  define PV_ASYNC 1
# endif
#endif

#if defined(PV_ASYNC)
typedef int pvi4 __attribute__((__vector_size__(4 * sizeof(int))));
typedef __attribute__((address_space(1))) pvi4 gvi4;   // global (prints as __device__)
typedef __attribute__((address_space(3))) pvi4 lvi4;   // LDS    (prints as __shared__)
#endif

namespace {
constexpr int kB = 8, kC = 64, kH = 256, kW = 256;
constexpr int BAND   = 8;            // output rows per workgroup
constexpr int NBANDS = kH / BAND;    // 32
constexpr int XTROWS = BAND + 4;     // 12 input rows (halo 2 for stacked 3x3)
constexpr int XTW    = 280;          // padded f32 tile width
constexpr int XOFF   = 12;           // logical col c -> phys c+12 (16B aligned)
constexpr int HROWS  = BAND + 2;     // 10 live h rows (halo 1)
constexpr int HRALLOC = HROWS + 1;   // +1 dump row (index 10) for clamped stores
constexpr int HW     = 280;
constexpr int HOFF   = 8;            // h logical col -8 -> phys 0
constexpr int NG1    = 17;           // conv1 col groups: x0 = -8 + 16g (h cols -8..263)
constexpr int NB1    = 3;            // conv1 row blocks of 4 (h rows -1..10, last 2 dumped)
constexpr int NG2    = 16;           // conv2 col groups: x0 = 16g
constexpr int NB2    = 2;            // conv2 row blocks of 4 (rows 0..7)
}

__global__ __launch_bounds__(256)
void pixvote_wmma_kernel(const float* __restrict__ x,
                         const float* __restrict__ w1, const float* __restrict__ b1,
                         const float* __restrict__ w2, const float* __restrict__ b2,
                         const float* __restrict__ wv, const float* __restrict__ bv,
                         float* __restrict__ out)
{
  __shared__ float    xt[XTROWS][XTW];     // input tile, f32, zero-padded borders
  __shared__ _Float16 ht[3][HRALLOC][HW];  // conv1 output (relu), f16 (+dump row)
  __shared__ float    acc[BAND][kW];       // channel-vote accumulator, f32

  const int  tid  = threadIdx.x;
  const int  lane = tid & 31;
  // Wave id is uniform across the wave: pin it to an SGPR so the task loops
  // below compile to scalar control flow (EXEC stays all-1s at the WMMAs).
  const int  wave = __builtin_amdgcn_readfirstlane(tid >> 5);   // 0..7
  const int  m    = lane & 15;             // A row index / B,D column index
  const bool hi   = lane >= 16;
  const int  band = blockIdx.x;
  const int  b    = blockIdx.y;
  const int  y0   = band * BAND;

  // ================= A matrices (contiguous k order, shared with B) ===============
  // lanes 0-15: elems 0..15 -> K0..15
  // lanes16-31: elems 0..15 -> K16..31
  //
  // conv1: M = oc*4 + sr (3 ch x 4 sub-rows), K = rr*3 + dx (6 rows x 3 dx)
  //        A1[m][k] = w1[oc][rr-sr][dx] when 0 <= rr-sr <= 2
  // conv2: M = sr (4 sub-rows), per-channel A2c[ch][m][k] = w2[ch][rr-m][dx]
  v16h a1, a2c0, a2c1, a2c2;
#pragma unroll
  for (int e = 0; e < 16; ++e) {
    const int k  = e + (hi ? 16 : 0);
    const int rr = k / 3, dx = k % 3;
    {
      const int oc = m >> 2, sr = m & 3, ky = rr - sr;
      a1[e] = (m < 12 && k < 18 && ky >= 0 && ky <= 2)
                  ? (_Float16)w1[oc * 9 + ky * 3 + dx] : (_Float16)0.0f;
    }
    {
      const int ky = rr - m;
      const bool ok = (m < 4 && k < 18 && ky >= 0 && ky <= 2);
      a2c0[e] = ok ? (_Float16)w2[0 * 9 + ky * 3 + dx] : (_Float16)0.0f;
      a2c1[e] = ok ? (_Float16)w2[1 * 9 + ky * 3 + dx] : (_Float16)0.0f;
      a2c2[e] = ok ? (_Float16)w2[2 * 9 + ky * 3 + dx] : (_Float16)0.0f;
    }
  }
  const float b1f0 = b1[0], b1f1 = b1[1], b1f2 = b1[2];
  const float b2s  = b2[0], bvs = bv[0];

  // ---- zero padded input tile (borders persist across channels) and acc ----
  for (int i = tid; i < XTROWS * XTW; i += 256) (&xt[0][0])[i]  = 0.0f;
  for (int i = tid; i < BAND * kW;   i += 256) (&acc[0][0])[i] = 0.0f;

  for (int c = 0; c < kC; ++c) {
    const float  wvc = wv[c];
    const float* xb  = x + ((size_t)(b * kC + c)) * kH * kW;

    __syncthreads();  // prior conv1/conv2 consumers done; xt safe to overwrite

    // ======== stage input rows y0-2 .. y0+9 into LDS (async if available) ===
    for (int i = tid; i < XTROWS * (kW / 4); i += 256) {
      const int rr = i >> 6, q = i & 63;
      const int ly = y0 - 2 + rr;
      if (0 <= ly && ly < kH) {
#if defined(PV_ASYNC)
        __builtin_amdgcn_global_load_async_to_lds_b128(
            (gvi4*)(xb + (size_t)ly * kW + 4 * q),
            (lvi4*)&xt[rr][XOFF + 4 * q], 0, 0);
#else
        const float4 v = reinterpret_cast<const float4*>(xb + (size_t)ly * kW)[q];
        *reinterpret_cast<float4*>(&xt[rr][XOFF + 4 * q]) = v;
#endif
      }
    }
#if defined(PV_ASYNC)
    __builtin_amdgcn_s_wait_asynccnt(0);
#endif
    __syncthreads();

    // ======== conv1: 1->3ch 3x3 + relu; one WMMA = 4 rows x 16 px x 3 ch ====
    // B k order here (contiguous, as A): lanes 0-15 hold K=0..15, lanes 16-31 hold K=16..31.
    // Branch-free gather: both half-wave tap paths are compile-time; merge
    // with a per-lane select (no EXEC manipulation around the WMMA).
    for (int t = wave; t < NB1 * NG1; t += 8) {
      const int bi = t / NG1;                 // row block: h rows 4bi-1 .. 4bi+2
      const int x0 = -8 + 16 * (t % NG1);
      const float* xr[6];
#pragma unroll
      for (int rr = 0; rr < 6; ++rr) {        // clamped input-row base pointers
        int row = 4 * bi + rr;                // (clamp only bites on dumped rows)
        row = row < XTROWS ? row : XTROWS - 1;
        xr[rr] = &xt[row][x0 + m - 1 + XOFF];
      }
      v16h bm;
#pragma unroll
      for (int e = 0; e < 16; ++e) {
        const float vl = xr[e / 3][e % 3];                                // k = e
        const float vh = (e < 2) ? xr[(e + 16) / 3][(e + 16) % 3] : 0.0f; // k = e+16
        bm[e] = (_Float16)(hi ? vh : vl);
      }
      const v8f cz = {0.f, 0.f, 0.f, 0.f, 0.f, 0.f, 0.f, 0.f};
      v8f d = __builtin_amdgcn_wmma_f32_16x16x32_f16(false, a1, false, bm,
                                                     (short)0, cz, false, false);
      // D: lanes 0-15 hold M=r (oc0,oc1 x 4 rows), lanes 16-31 hold M=r+8 (oc2)
      // Out-of-band rows (block 2, sr=2,3) are clamped into the dump row 10:
      // stores are unconditional, no exec-mask churn.
      const int col = x0 + m + HOFF;
      if (!hi) {
#pragma unroll
        for (int r = 0; r < 8; ++r) {
          const int oc = r >> 2, sr = r & 3;
          int hr = 4 * bi + sr; hr = hr < HROWS ? hr : HROWS;   // 10 = dump row
          const float bias = (oc == 0) ? b1f0 : b1f1;
          ht[oc][hr][col] = (_Float16)fmaxf(d[r] + bias, 0.0f);
        }
      } else {
#pragma unroll
        for (int r = 0; r < 4; ++r) {
          int hr = 4 * bi + r; hr = hr < HROWS ? hr : HROWS;
          ht[2][hr][col] = (_Float16)fmaxf(d[r] + b1f2, 0.0f);
        }
      }
    }
    __syncthreads();

    // ======== conv2: 3->1ch 3x3 + sigmoid + vote; 3 chained WMMAs ===========
    // One task = 4 output rows x 16 px; C-operand chains the channel sum.
    for (int t = wave; t < NB2 * NG2; t += 8) {
      const int bj = t >> 4;                  // rows 4bj .. 4bj+3
      const int x0 = (t & 15) << 4;
      v8f d = {0.f, 0.f, 0.f, 0.f, 0.f, 0.f, 0.f, 0.f};
#pragma unroll
      for (int ch = 0; ch < 3; ++ch) {
        const _Float16* hr_[6];
#pragma unroll
        for (int rr = 0; rr < 6; ++rr)
          hr_[rr] = &ht[ch][4 * bj + rr][x0 + m - 1 + HOFF];
        v16h bm;
#pragma unroll
        for (int e = 0; e < 16; ++e) {
          const _Float16 vl = hr_[e / 3][e % 3];
          const _Float16 vh = (e < 2) ? hr_[(e + 16) / 3][(e + 16) % 3]
                                      : (_Float16)0.0f;
          bm[e] = hi ? vh : vl;
        }
        const v16h a2 = (ch == 0) ? a2c0 : (ch == 1) ? a2c1 : a2c2;
        d = __builtin_amdgcn_wmma_f32_16x16x32_f16(false, a2, false, bm,
                                                   (short)0, d, false, false);
      }
      if (!hi) {                              // D rows 0..3 live in lanes 0-15
#pragma unroll
        for (int r = 0; r < 4; ++r) {
          const float s = 1.0f / (1.0f + __expf(-(d[r] + b2s)));
          // one writer per element per channel iteration -> ds_add_f32,
          // sequential channel order keeps the sum deterministic
          atomicAdd(&acc[4 * bj + r][x0 + m], wvc * s);
        }
      }
    }
  }

  __syncthreads();
#pragma unroll
  for (int r = 0; r < BAND; ++r)
    out[((size_t)b * kH + (y0 + r)) * kW + tid] = acc[r][tid] + bvs;
}

extern "C" void kernel_launch(void* const* d_in, const int* in_sizes, int n_in,
                              void* d_out, int out_size, void* d_ws, size_t ws_size,
                              hipStream_t stream) {
  const float* x  = (const float*)d_in[0];
  const float* w1 = (const float*)d_in[1];
  const float* b1 = (const float*)d_in[2];
  const float* w2 = (const float*)d_in[3];
  const float* b2 = (const float*)d_in[4];
  const float* wv = (const float*)d_in[5];
  const float* bv = (const float*)d_in[6];
  float* out = (float*)d_out;

  dim3 grid(NBANDS, kB);   // 32 row-bands x 8 batches = 256 workgroups
  pixvote_wmma_kernel<<<grid, 256, 0, stream>>>(x, w1, b1, w2, b2, wv, bv, out);
}
